// Bottleneck_7911329759669
// MI455X (gfx1250) — matrix operand order repaired and checked by reading, not compiled, not run on hardware
//
#include <hip/hip_runtime.h>
#include <math.h>

typedef __attribute__((ext_vector_type(16))) __bf16 v16bf;
typedef __attribute__((ext_vector_type(8)))  __bf16 v8bf;
typedef __attribute__((ext_vector_type(4)))  __bf16 v4bf;
typedef __attribute__((ext_vector_type(8)))  float  v8f;
typedef __attribute__((ext_vector_type(4)))  unsigned int v4u;
typedef __attribute__((ext_vector_type(4)))  int v4i;
typedef __attribute__((ext_vector_type(8)))  int v8i;

namespace {
constexpr int kB    = 64;
constexpr int kCin  = 256;
constexpr int kH    = 56;
constexpr int kW    = 56;
constexpr int kHW   = kH * kW;       // 3136
constexpr int kWid  = 64;
constexpr int kCout = 256;
constexpr int kKsp  = 1568;          // spatial keep
constexpr int kKch  = 32;            // channel keep
constexpr int kM    = kB * kHW;      // 200704 pixels
constexpr float kEps = 1e-5f;
// 40 halves = 80B row stride: multiple of 16B (aligned b128 LDS stores) and
// conflict-free fragment reads (20 dwords/row, 16 distinct banks over m=0..15)
constexpr int APAD = 40;
constexpr int BPAD = 40;
}

#if __has_builtin(__builtin_amdgcn_tensor_load_to_lds) && \
    __has_builtin(__builtin_amdgcn_s_wait_tensorcnt)
#define USE_TDM 1
#warning "CDNA5 probe: TDM path ON"
#else
#define USE_TDM 0
#warning "CDNA5 probe: TDM path OFF"
#endif

#if __has_builtin(__builtin_amdgcn_global_load_async_to_lds_b128) && \
    __has_builtin(__builtin_amdgcn_s_wait_asynccnt)
#define USE_ASYNC 1
#warning "CDNA5 probe: async-to-LDS path ON"
#else
#define USE_ASYNC 0
#warning "CDNA5 probe: async-to-LDS path OFF"
#endif

// Pointer types matching the builtin's signature (diagnostic showed
// parameter 1 as 'v4i __device__*', i.e. int32x4 in target AS 1).
typedef __attribute__((address_space(1))) v4i* as1_v4i_p;
typedef __attribute__((address_space(3))) v4i* as3_v4i_p;

// ---------------- WMMA fragment loads (CDNA5 wave32 layouts, CDNA5 ISA) -----
__device__ __forceinline__ v16bf frag_a(const __bf16* lds, int lane) {
  int m  = lane & 15;
  int kb = (lane >> 4) << 4;
  v16bf a;
#pragma unroll
  for (int v = 0; v < 8; ++v) {
    int k = kb + (v << 1);
    a[2 * v]     = lds[m * APAD + k];
    a[2 * v + 1] = lds[m * APAD + k + 1];
  }
  return a;
}
__device__ __forceinline__ v16bf frag_b(const __bf16* lds, int lane, int ntile) {
  int n  = (lane & 15) + ntile * 16;
  int kb = (lane >> 4) << 4;
  v16bf b;
#pragma unroll
  for (int v = 0; v < 8; ++v) {
    int k = kb + (v << 1);
    b[2 * v]     = lds[n * BPAD + k];
    b[2 * v + 1] = lds[n * BPAD + k + 1];
  }
  return b;
}

__device__ __forceinline__ v8bf zero8bf() {
  v8bf z;
#pragma unroll
  for (int i = 0; i < 8; ++i) z[i] = (__bf16)0.0f;
  return z;
}

// ---------------- masks kernel -----------------------------------------------
__global__ __launch_bounds__(256) void k_masks(
    const float* __restrict__ x, const float* __restrict__ fcw,
    const float* __restrict__ fcb, const float* __restrict__ mw,
    const float* __restrict__ mb, float* __restrict__ vec,
    float* __restrict__ mask, float* __restrict__ maskd) {
  __shared__ float pooled[kCin];
  __shared__ float sal[kWid];
  __shared__ float sp[kHW];
  __shared__ float mwl[kCin];
  __shared__ unsigned cntS;
  __shared__ float thrS;

  const int bIdx = blockIdx.x, t = threadIdx.x;
  const int lane = t & 31, wv = t >> 5;
  const float* xb = x + (size_t)bIdx * kCin * kHW;

  for (int c = wv; c < kCin; c += 8) {
    float s = 0.f;
    for (int hw = lane; hw < kHW; hw += 32) s += xb[(size_t)c * kHW + hw];
#pragma unroll
    for (int o = 16; o; o >>= 1) s += __shfl_xor(s, o, 32);
    if (lane == 0) pooled[c] = s * (1.0f / kHW);
  }
  if (t < kCin) mwl[t] = mw[t];
  __syncthreads();

  if (t < kWid) {
    float a = fcb[t];
    for (int c = 0; c < kCin; ++c) a += pooled[c] * fcw[t * kCin + c];
    sal[t] = 1.0f / (1.0f + expf(-a));
  }
  __syncthreads();
  if (t == 0) {  // exact k-th largest of 64 incl. ties (matches top_k + >=)
    float T = -1e30f;
    for (int n = 0; n < kWid; ++n) {
      int gt = 0, ge = 0; float v = sal[n];
      for (int m2 = 0; m2 < kWid; ++m2) { gt += sal[m2] > v; ge += sal[m2] >= v; }
      if (gt < kKch && ge >= kKch) T = v;
    }
    thrS = T;
  }
  __syncthreads();
  if (t < kWid) vec[bIdx * kWid + t] = (sal[t] >= thrS) ? 1.0f : 0.0f;

  for (int hw = t; hw < kHW; hw += 256) {
    float a = mb[0];
    for (int c = 0; c < kCin; ++c) a += xb[(size_t)c * kHW + hw] * mwl[c];
    sp[hw] = a;
  }
  __syncthreads();

  unsigned keys[13]; int nk = 0;
  for (int hw = t; hw < kHW; hw += 256) {
    unsigned u = __float_as_uint(sp[hw]);
    keys[nk++] = (u & 0x80000000u) ? ~u : (u | 0x80000000u);
  }
  unsigned long long lo = 0, hi = 0x100000000ull;
  while (hi - lo > 1) {
    unsigned mid = (unsigned)((lo + hi) >> 1);
    if (t == 0) cntS = 0;
    __syncthreads();
    unsigned c = 0;
    for (int i = 0; i < nk; ++i) c += (keys[i] >= mid);
    atomicAdd(&cntS, c);
    __syncthreads();
    if (cntS >= (unsigned)kKsp) lo = mid; else hi = mid;
    __syncthreads();
  }
  const unsigned T = (unsigned)lo;

  { int i = 0;
    for (int hw = t; hw < kHW; hw += 256) sp[hw] = (keys[i++] >= T) ? 1.f : 0.f; }
  __syncthreads();
  for (int hw = t; hw < kHW; hw += 256) {
    mask[bIdx * kHW + hw] = sp[hw];
    int h = hw / kW, w = hw % kW;
    float md = 0.f;
    for (int dy = -1; dy <= 1; ++dy)
      for (int dx = -1; dx <= 1; ++dx) {
        int h2 = h + dy, w2 = w + dx;
        if ((unsigned)h2 < (unsigned)kH && (unsigned)w2 < (unsigned)kW)
          md = fmaxf(md, sp[h2 * kW + w2]);
      }
    maskd[bIdx * kHW + hw] = md;
  }
}

// ---------------- conv1: 1x1 GEMM [M,256]x[256,64] ---------------------------
__global__ __launch_bounds__(256) void k_conv1(
    const float* __restrict__ x, const float* __restrict__ w1,
    const float* __restrict__ g, const float* __restrict__ bta,
    const float* __restrict__ mu, const float* __restrict__ va,
    const float* __restrict__ maskd, const float* __restrict__ vec,
    __bf16* __restrict__ act1) {
  __shared__ __bf16 sA[128 * APAD];
  __shared__ __bf16 sB[64 * BPAD];
  const int t = threadIdx.x, lane = t & 31, wv = t >> 5;
  const int m0 = blockIdx.x * 128;
  v8f acc[4] = {};
  for (int ks = 0; ks < kCin; ks += 32) {
    __syncthreads();
#pragma unroll
    for (int i = 0; i < 4; ++i) {          // A: float4 along pixels (16B aligned)
      int e = i * 256 + t;
      int c = e >> 5;
      int pq = (e & 31) << 2;
      int p = m0 + pq;
      int batch = p / kHW, hw = p % kHW;
      const float4 v = *(const float4*)&x[(size_t)batch * kCin * kHW +
                                          (size_t)(ks + c) * kHW + hw];
      sA[(pq + 0) * APAD + c] = (__bf16)v.x;
      sA[(pq + 1) * APAD + c] = (__bf16)v.y;
      sA[(pq + 2) * APAD + c] = (__bf16)v.z;
      sA[(pq + 3) * APAD + c] = (__bf16)v.w;
    }
#pragma unroll
    for (int i = 0; i < 2; ++i) {          // B: w1[n][k], float4 -> packed v4bf
      int e = i * 256 + t;
      int n = e >> 3, kq = (e & 7) << 2;
      const float4 v = *(const float4*)&w1[n * kCin + ks + kq];
      v4bf b4 = {(__bf16)v.x, (__bf16)v.y, (__bf16)v.z, (__bf16)v.w};
      *(v4bf*)&sB[n * BPAD + kq] = b4;
    }
    __syncthreads();
    v16bf af = frag_a(sA + wv * 16 * APAD, lane);
    v16bf bfr[4];
#pragma unroll
    for (int nt = 0; nt < 4; ++nt) bfr[nt] = frag_b(sB, lane, nt);
#pragma unroll
    for (int nt = 0; nt < 4; ++nt)
      acc[nt] = __builtin_amdgcn_wmma_f32_16x16x32_bf16(
          false, af, false, bfr[nt], (short)0, acc[nt], false, false);
  }
  const int nlo = lane & 15, mrow = (lane >> 4) * 8;
#pragma unroll
  for (int nt = 0; nt < 4; ++nt) {
    int n = nt * 16 + nlo;
    float sc = g[n] * rsqrtf(va[n] + kEps);
    float bi = bta[n] - mu[n] * sc;
    float vv_ = 0.f;
#pragma unroll
    for (int r = 0; r < 8; ++r) {
      int p = m0 + wv * 16 + mrow + r;
      int batch = p / kHW;
      if (r == 0) vv_ = vec[batch * kWid + n];
      float y = fmaxf(acc[nt][r] * sc + bi, 0.f) * maskd[p] * vv_;
      act1[(size_t)p * kWid + n] = (__bf16)y;
    }
  }
}

// ---------------- conv2: 3x3 im2col GEMM, K = tap*64+cin ---------------------
__global__ __launch_bounds__(256) void k_conv2(
    const __bf16* __restrict__ act1, const float* __restrict__ w2,
    const float* __restrict__ g, const float* __restrict__ bta,
    const float* __restrict__ mu, const float* __restrict__ va,
    const float* __restrict__ mask, const float* __restrict__ vec,
    __bf16* __restrict__ act2) {
  __shared__ __bf16 sA[128 * APAD];
  __shared__ __bf16 sB[64 * BPAD];
  const int t = threadIdx.x, lane = t & 31, wv = t >> 5;
  const int m0 = blockIdx.x * 128;
  v8f acc[4] = {};
  for (int ks = 0; ks < kWid * 9; ks += 32) {
    const int off = ks >> 6;
    const int cinb = ks & 63;
    const int dy = off / 3 - 1, dx = off % 3 - 1;
    __syncthreads();
#pragma unroll
    for (int i = 0; i < 2; ++i) {           // A: per-lane gather -> LDS
      int e = i * 256 + t;
      int pl = e >> 2, co = (e & 3) << 3;
      int p = m0 + pl;
      int batch = p / kHW, hw = p % kHW;
      int h = hw / kW, w = hw % kW;
      int h2 = h + dy, w2 = w + dx;
      bool ok = (unsigned)h2 < (unsigned)kH && (unsigned)w2 < (unsigned)kW;
      __bf16* ldst = &sA[pl * APAD + co];
#if USE_ASYNC
      if (ok) {
        const __bf16* gsrc =
            &act1[((size_t)batch * kHW + h2 * kW + w2) * kWid + cinb + co];
        __builtin_amdgcn_global_load_async_to_lds_b128(
            (as1_v4i_p)(unsigned long long)(size_t)gsrc,
            (as3_v4i_p)(unsigned)(size_t)ldst, 0, 0);
      } else {
        *(v8bf*)ldst = zero8bf();
      }
#else
      v8bf val = zero8bf();
      if (ok)
        val = *(const v8bf*)&act1[((size_t)batch * kHW + h2 * kW + w2) * kWid +
                                  cinb + co];
      *(v8bf*)ldst = val;
#endif
    }
#pragma unroll
    for (int i = 0; i < 8; ++i) {           // B: stride-9 reorder (scalar gather)
      int idx = i * 256 + t;
      int n = idx >> 5, k = idx & 31;
      sB[n * BPAD + k] = (__bf16)w2[n * (kWid * 9) + (cinb + k) * 9 + off];
    }
#if USE_ASYNC
    __builtin_amdgcn_s_wait_asynccnt(0);
#endif
    __syncthreads();
    v16bf af = frag_a(sA + wv * 16 * APAD, lane);
    v16bf bfr[4];
#pragma unroll
    for (int nt = 0; nt < 4; ++nt) bfr[nt] = frag_b(sB, lane, nt);
#pragma unroll
    for (int nt = 0; nt < 4; ++nt)
      acc[nt] = __builtin_amdgcn_wmma_f32_16x16x32_bf16(
          false, af, false, bfr[nt], (short)0, acc[nt], false, false);
  }
  const int nlo = lane & 15, mrow = (lane >> 4) * 8;
#pragma unroll
  for (int nt = 0; nt < 4; ++nt) {
    int n = nt * 16 + nlo;
    float sc = g[n] * rsqrtf(va[n] + kEps);
    float bi = bta[n] - mu[n] * sc;
    float vv_ = 0.f;
#pragma unroll
    for (int r = 0; r < 8; ++r) {
      int p = m0 + wv * 16 + mrow + r;
      int batch = p / kHW;
      if (r == 0) vv_ = vec[batch * kWid + n];
      float y = fmaxf(acc[nt][r] * sc + bi, 0.f) * mask[p] * vv_;
      act2[(size_t)p * kWid + n] = (__bf16)y;
    }
  }
}

// ---------------- conv3: 1x1 GEMM [M,64]x[64,256] + identity -----------------
__global__ __launch_bounds__(256) void k_conv3(
    const __bf16* __restrict__ act2, const float* __restrict__ w3,
    const float* __restrict__ g, const float* __restrict__ bta,
    const float* __restrict__ mu, const float* __restrict__ va,
    const float* __restrict__ mask, const float* __restrict__ x,
    float* __restrict__ out) {
  __shared__ __bf16 sA[128 * APAD];
  __shared__ __bf16 sB[64 * BPAD];
  __shared__ float sOut[128 * 65];
  __shared__ float sc3[64], bi3[64];
  const int t = threadIdx.x, lane = t & 31, wv = t >> 5;
  const int m0 = blockIdx.x * 128;
  const int nBase = blockIdx.y * 64;
  if (t < 64) {
    int ng = nBase + t;
    float sc = g[ng] * rsqrtf(va[ng] + kEps);
    sc3[t] = sc;
    bi3[t] = bta[ng] - mu[ng] * sc;
  }
  v8f acc[4] = {};
  for (int ks = 0; ks < kWid; ks += 32) {
    __syncthreads();
#if USE_TDM
    // TDM: one 128x32 bf16 tile, row stride 64 elems, LDS rows padded 32->40
    // halves via pad_interval=16 dwords / pad_amount=4 dwords.
    if (wv == 0) {
      unsigned ldsOff = (unsigned)(size_t)(void*)sA;
      unsigned long long ga =
          (unsigned long long)(size_t)(act2 + (size_t)m0 * kWid + ks);
      v4u g0;
      g0[0] = 1u;
      g0[1] = ldsOff;
      g0[2] = (unsigned)(ga & 0xffffffffu);
      g0[3] = (unsigned)((ga >> 32) & 0x01ffffffu) | 0x80000000u;  // type=2
      v8i g1;
      g1[0] = (1 << 16) | (1 << 20) | (3 << 22) | (3 << 25);
      g1[1] = (kWid << 16);
      g1[2] = (int)((unsigned)(kM & 0xffff) << 16);
      g1[3] = (kM >> 16) | (32 << 16);
      g1[4] = 128;
      g1[5] = kWid;
      g1[6] = 0;
      g1[7] = 0;
      v4i g2 = {1, 1, 0, 0};
      v4i g3 = {0, 1 << 16, 0, 0};
#if defined(__clang_major__) && (__clang_major__ >= 23)
      v8i g4 = {0, 0, 0, 0, 0, 0, 0, 0};
      __builtin_amdgcn_tensor_load_to_lds(g0, g1, g2, g3, g4, 0);
#else
      __builtin_amdgcn_tensor_load_to_lds(g0, g1, g2, g3, 0);
#endif
    }
#else
#pragma unroll
    for (int i = 0; i < 2; ++i) {
      int e = i * 256 + t;
      int pl = e >> 2, co = (e & 3) << 3;
      *(v8bf*)&sA[pl * APAD + co] =
          *(const v8bf*)&act2[(size_t)(m0 + pl) * kWid + ks + co];
    }
#endif
#pragma unroll
    for (int i = 0; i < 2; ++i) {           // B: w3 float4 -> packed v4bf
      int e = i * 256 + t;
      int n = e >> 3, kq = (e & 7) << 2;
      const float4 v = *(const float4*)&w3[(size_t)(nBase + n) * kWid + ks + kq];
      v4bf b4 = {(__bf16)v.x, (__bf16)v.y, (__bf16)v.z, (__bf16)v.w};
      *(v4bf*)&sB[n * BPAD + kq] = b4;
    }
#if USE_TDM
    if (wv == 0) __builtin_amdgcn_s_wait_tensorcnt(0);
#endif
    __syncthreads();
    v16bf af = frag_a(sA + wv * 16 * APAD, lane);
    v16bf bfr[4];
#pragma unroll
    for (int nt = 0; nt < 4; ++nt) bfr[nt] = frag_b(sB, lane, nt);
#pragma unroll
    for (int nt = 0; nt < 4; ++nt)
      acc[nt] = __builtin_amdgcn_wmma_f32_16x16x32_bf16(
          false, af, false, bfr[nt], (short)0, acc[nt], false, false);
  }
  const int nlo = lane & 15, mrow = (lane >> 4) * 8;
#pragma unroll
  for (int nt = 0; nt < 4; ++nt)
#pragma unroll
    for (int r = 0; r < 8; ++r)
      sOut[(wv * 16 + mrow + r) * 65 + nt * 16 + nlo] = acc[nt][r];
  __syncthreads();
#pragma unroll
  for (int i = 0; i < 32; ++i) {            // coalesced NCHW epilogue
    int idx = i * 256 + t;
    int n = idx >> 7, ml = idx & 127;
    int p = m0 + ml;
    int batch = p / kHW, hw = p % kHW;
    int ng = nBase + n;
    size_t gi = (size_t)batch * kCout * kHW + (size_t)ng * kHW + hw;
    float y = sOut[ml * 65 + n] * sc3[n] + bi3[n];
    y = y * mask[p] + x[gi];
    out[gi] = fmaxf(y, 0.f);
  }
}

extern "C" void kernel_launch(void* const* d_in, const int* in_sizes, int n_in,
                              void* d_out, int out_size, void* d_ws, size_t ws_size,
                              hipStream_t stream) {
  (void)in_sizes; (void)n_in; (void)out_size; (void)ws_size;
  const float* x   = (const float*)d_in[0];
  const float* w1  = (const float*)d_in[1];
  const float* g1  = (const float*)d_in[2];
  const float* b1  = (const float*)d_in[3];
  const float* m1  = (const float*)d_in[4];
  const float* v1  = (const float*)d_in[5];
  const float* w2  = (const float*)d_in[6];
  const float* g2  = (const float*)d_in[7];
  const float* b2  = (const float*)d_in[8];
  const float* m2  = (const float*)d_in[9];
  const float* v2  = (const float*)d_in[10];
  const float* w3  = (const float*)d_in[11];
  const float* g3  = (const float*)d_in[12];
  const float* b3  = (const float*)d_in[13];
  const float* m3  = (const float*)d_in[14];
  const float* v3  = (const float*)d_in[15];
  const float* fcw = (const float*)d_in[16];
  const float* fcb = (const float*)d_in[17];
  const float* mw  = (const float*)d_in[18];
  const float* mb  = (const float*)d_in[19];

  char* ws = (char*)d_ws;
  size_t off = 0;
  float*  vecP  = (float*)(ws + off); off += (size_t)kB * kWid * sizeof(float);
  float*  maskP = (float*)(ws + off); off += (size_t)kB * kHW * sizeof(float);
  float*  maskD = (float*)(ws + off); off += (size_t)kB * kHW * sizeof(float);
  __bf16* act1  = (__bf16*)(ws + off); off += (size_t)kM * kWid * sizeof(__bf16);
  __bf16* act2  = (__bf16*)(ws + off);

  k_masks<<<dim3(kB), dim3(256), 0, stream>>>(x, fcw, fcb, mw, mb, vecP, maskP, maskD);
  k_conv1<<<dim3(kM / 128), dim3(256), 0, stream>>>(x, w1, g1, b1, m1, v1, maskD, vecP, act1);
  k_conv2<<<dim3(kM / 128), dim3(256), 0, stream>>>(act1, w2, g2, b2, m2, v2, maskP, vecP, act2);
  k_conv3<<<dim3(kM / 128, kCout / 64), dim3(256), 0, stream>>>(
      act2, w3, g3, b3, m3, v3, maskP, x, (float*)d_out);
}
